// BinaryConv2DLayer_38225208935077
// MI455X (gfx1250) — compile-verified, matrix operand order checked by reading, not run on hardware
//
#include <hip/hip_runtime.h>

typedef __attribute__((ext_vector_type(16))) _Float16 v16h;
typedef __attribute__((ext_vector_type(8)))  _Float16 v8h;
typedef __attribute__((ext_vector_type(8)))  float    v8f;
typedef __attribute__((ext_vector_type(4)))  float    v4f;

#define IMG_H   56
#define IMG_W   56
#define CIN     256
#define COUT    256
#define BM      128
#define BN      128
#define BK      32
#define KBLOCKS 72        // 3*3*256 / 32
#define KSTRIDE 40        // padded LDS row stride (halves): 80B rows -> conflict-free b128 reads

// Implicit-GEMM binary-weight conv: D = im2col(x) * sign(W), f16 WMMA, fp32 accum.
// Block tile 128x128, K-step 32; 8 waves in 4x2, wave tile 32x64 (2x4 WMMA per K-step).
__global__ __launch_bounds__(256) void bconv2d_wmma(const float* __restrict__ x,
                                                    const float* __restrict__ w,
                                                    float* __restrict__ out) {
  __shared__ _Float16 As[2][BM][KSTRIDE];   // 20,480 B
  __shared__ _Float16 Bs[2][BN][KSTRIDE];   // 20,480 B

  const int tid  = threadIdx.x;
  const int m0   = blockIdx.y * BM;         // flat (n,h,w) row base
  const int oc0  = blockIdx.x * BN;         // output-channel base

  const int wave = tid >> 5;
  const int lane = tid & 31;
  const int l16  = lane & 15;
  const int half = lane >> 4;
  const int wm   = (wave & 3) * 32;         // wave tile: 32 rows x 64 cols
  const int wn   = (wave >> 2) * 64;

  // ---- A-tile loader mapping: 2 threads per row, 16 floats each ----
  const int arow = tid >> 1;
  const int acol = (tid & 1) * 16;
  const int am   = m0 + arow;
  const int an   = am / (IMG_H * IMG_W);
  const int ahw  = am % (IMG_H * IMG_W);
  const int ah   = ahw / IMG_W;
  const int aw_  = ahw % IMG_W;

  // Hoisted im2col state: one base pointer + 9-bit (dh,dw) validity mask.
  const float* abase =
      x + (long)((an * IMG_H + ah - 1) * IMG_W + (aw_ - 1)) * CIN + acol;
  int amask = 0;
#pragma unroll
  for (int dh = 0; dh < 3; ++dh)
#pragma unroll
    for (int dw = 0; dw < 3; ++dw) {
      const int ih = ah + dh - 1, iw = aw_ + dw - 1;
      if ((unsigned)ih < IMG_H && (unsigned)iw < IMG_W) amask |= 1 << (dh * 3 + dw);
    }

  // ---- B-tile loader mapping: one oc row per thread-mod-128, 16 k's each ----
  const int boc = tid & 127;                // 0..127
  const int bk0 = (tid >> 7) * 16;          // 0 or 16

  v8f acc[2][4] = {};

  // Uniform K-block decode -> (gather offset, mask bit position).
  auto aOff = [&](int kb, int& pos) -> int {
    const int dh = kb / 24;
    const int rm = kb % 24;
    const int dw = rm >> 3;
    const int c0 = (rm & 7) * 32;
    pos = dh * 3 + dw;
    return dh * (IMG_W * CIN) + dw * CIN + c0;   // wave-uniform
  };

  auto loadA = [&](int kb, int buf) {
    int pos;
    const int off = aOff(kb, pos);
    const bool inb = (amask >> pos) & 1;
    const float* src = abase + off;
    v8h lo, hi;
    if (inb) {
      v4f v0 = *(const v4f*)(src);
      v4f v1 = *(const v4f*)(src + 4);
      v4f v2 = *(const v4f*)(src + 8);
      v4f v3 = *(const v4f*)(src + 12);
#if __has_builtin(__builtin_amdgcn_cvt_pkrtz)
      auto p0 = __builtin_amdgcn_cvt_pkrtz(v0[0], v0[1]);
      auto p1 = __builtin_amdgcn_cvt_pkrtz(v0[2], v0[3]);
      auto p2 = __builtin_amdgcn_cvt_pkrtz(v1[0], v1[1]);
      auto p3 = __builtin_amdgcn_cvt_pkrtz(v1[2], v1[3]);
      auto p4 = __builtin_amdgcn_cvt_pkrtz(v2[0], v2[1]);
      auto p5 = __builtin_amdgcn_cvt_pkrtz(v2[2], v2[3]);
      auto p6 = __builtin_amdgcn_cvt_pkrtz(v3[0], v3[1]);
      auto p7 = __builtin_amdgcn_cvt_pkrtz(v3[2], v3[3]);
      lo[0] = p0[0]; lo[1] = p0[1]; lo[2] = p1[0]; lo[3] = p1[1];
      lo[4] = p2[0]; lo[5] = p2[1]; lo[6] = p3[0]; lo[7] = p3[1];
      hi[0] = p4[0]; hi[1] = p4[1]; hi[2] = p5[0]; hi[3] = p5[1];
      hi[4] = p6[0]; hi[5] = p6[1]; hi[6] = p7[0]; hi[7] = p7[1];
#else
#pragma unroll
      for (int e = 0; e < 4; ++e) { lo[e] = (_Float16)v0[e]; lo[4 + e] = (_Float16)v1[e]; }
#pragma unroll
      for (int e = 0; e < 4; ++e) { hi[e] = (_Float16)v2[e]; hi[4 + e] = (_Float16)v3[e]; }
#endif
    } else {
#pragma unroll
      for (int e = 0; e < 8; ++e) { lo[e] = (_Float16)0.0f; hi[e] = (_Float16)0.0f; }
    }
    *(v8h*)&As[buf][arow][acol]     = lo;
    *(v8h*)&As[buf][arow][acol + 8] = hi;
  };

  auto loadB = [&](int kb, int buf) {
    // HWIO flat k = kb*32 + bk0 + i; W[k][oc] -> +/-1 in f16 (exact).
    const float* src = w + (size_t)(kb * BK + bk0) * COUT + oc0 + boc;
    v8h t0, t1;
#pragma unroll
    for (int i = 0; i < 8; ++i)
      t0[i] = (src[i * COUT] >= 0.0f) ? (_Float16)1.0f : (_Float16)-1.0f;
#pragma unroll
    for (int i = 0; i < 8; ++i)
      t1[i] = (src[(8 + i) * COUT] >= 0.0f) ? (_Float16)1.0f : (_Float16)-1.0f;
    *(v8h*)&Bs[buf][boc][bk0]     = t0;
    *(v8h*)&Bs[buf][boc][bk0 + 8] = t1;
  };

  loadA(0, 0);
  loadB(0, 0);
  __syncthreads();

  for (int kb = 0; kb < KBLOCKS; ++kb) {
    const int cur = kb & 1;
    if (kb + 1 < KBLOCKS) {                 // stage next tile into the other buffer
      loadA(kb + 1, cur ^ 1);
      loadB(kb + 1, cur ^ 1);
    }
    if (kb + 2 < KBLOCKS) {                 // warm caches ahead of the double buffer
      int pos;
      const int off = aOff(kb + 2, pos);
      if ((amask >> pos) & 1) __builtin_prefetch(abase + off, 0, 1);
    }

    // Fragment loads; LDS layout matches 16-bit operand striping:
    // lane l = row, elems 0..7 -> K = half*8 + e, elems 8..15 -> K = 16 + half*8 + e.
    v16h afrag[2], bfrag[4];
#pragma unroll
    for (int i = 0; i < 2; ++i) {
      const _Float16* ap = &As[cur][wm + i * 16 + l16][half * 8];
      v8h lo = *(const v8h*)ap;
      v8h hi = *(const v8h*)(ap + 16);
#pragma unroll
      for (int e = 0; e < 8; ++e) { afrag[i][e] = lo[e]; afrag[i][8 + e] = hi[e]; }
    }
#pragma unroll
    for (int j = 0; j < 4; ++j) {
      const _Float16* bp = &Bs[cur][wn + j * 16 + l16][half * 8];
      v8h lo = *(const v8h*)bp;
      v8h hi = *(const v8h*)(bp + 16);
#pragma unroll
      for (int e = 0; e < 8; ++e) { bfrag[j][e] = lo[e]; bfrag[j][8 + e] = hi[e]; }
    }

#pragma unroll
    for (int i = 0; i < 2; ++i)
#pragma unroll
      for (int j = 0; j < 4; ++j)
        acc[i][j] = __builtin_amdgcn_wmma_f32_16x16x32_f16(
            false, afrag[i], false, bfrag[j], (short)0, acc[i][j], false, false);

    __syncthreads();                        // readers done with 'cur' before it is re-filled
  }

  // C/D layout: VGPR e, lanes 0-15 -> M = e (+i*16), lanes 16-31 -> M = e+8; N = l16.
#pragma unroll
  for (int i = 0; i < 2; ++i) {
#pragma unroll
    for (int j = 0; j < 4; ++j) {
      const int row0 = m0 + wm + i * 16 + half * 8;
      const int col  = oc0 + wn + j * 16 + l16;
#pragma unroll
      for (int e = 0; e < 8; ++e)
        out[(size_t)(row0 + e) * COUT + col] = acc[i][j][e];
    }
  }
}

extern "C" void kernel_launch(void* const* d_in, const int* in_sizes, int n_in,
                              void* d_out, int out_size, void* d_ws, size_t ws_size,
                              hipStream_t stream) {
  (void)in_sizes; (void)n_in; (void)out_size; (void)d_ws; (void)ws_size;
  const float* x = (const float*)d_in[0];   // (32,56,56,256) fp32 NHWC
  const float* w = (const float*)d_in[1];   // (3,3,256,256) fp32 HWIO
  float* out = (float*)d_out;               // (32,56,56,256) fp32 NHWC

  dim3 grid(COUT / BN, (32 * IMG_H * IMG_W) / BM);  // (2, 784)
  bconv2d_wmma<<<grid, 256, 0, stream>>>(x, w, out);
}
